// FAquantizer_30923764531362
// MI455X (gfx1250) — compile-verified, matrix operand order checked by reading, not run on hardware
//
#include <hip/hip_runtime.h>
#include <hip/hip_bf16.h>
#include <math.h>

typedef _Float16 h16;
typedef _Float16 v16h __attribute__((ext_vector_type(16)));
typedef _Float16 v8h  __attribute__((ext_vector_type(8)));
typedef float    v8f  __attribute__((ext_vector_type(8)));

#define BB 16
#define TT 1024
#define DD 1024
#define HH 256
#define BT (BB*TT)          // 16384 rows
#define CBK 1024
#define CBD 8

// ---------------------------------------------------------------------------
// NT GEMM:  C[M,N] = A[M,K] (f16, row-major, lda) * Bt[N,K]^T (f16, row-major, ldb)
//           + bias[N], f32 accumulate via V_WMMA_F32_16X16X32_F16.
// Block = 256 threads = 8 waves; wave tile 64x32 (4x2 wmma tiles); K step 32.
// Row loads clamped (buffers padded), stores masked by (r<M, c<N).
// ep: 0 = bias only, 1 = leaky_relu(0.2) after bias.
// ---------------------------------------------------------------------------
__global__ __launch_bounds__(256) void gemm_wmma_nt(
    const h16* __restrict__ A, int lda, int aclamp,
    const h16* __restrict__ Bt, int ldb, int bclamp,
    const float* __restrict__ bias,
    float* __restrict__ C, int ldc,
    int M, int N, int K, int ep)
{
  const int wid  = threadIdx.x >> 5;
  const int lane = threadIdx.x & 31;
  const int wm = wid >> 2;            // 0..1
  const int wn = wid & 3;             // 0..3
  const int m0 = blockIdx.y * 128 + wm * 64;
  const int n0 = blockIdx.x * 128 + wn * 32;
  const int sub = lane & 15;
  const int hi  = lane >> 4;
  const int koff = hi ? 8 : 0;        // lanes 16..31 hold K+8 / K+24 packs

  v8f acc[4][2];
#pragma unroll
  for (int mi = 0; mi < 4; ++mi)
#pragma unroll
    for (int ni = 0; ni < 2; ++ni)
#pragma unroll
      for (int v = 0; v < 8; ++v) acc[mi][ni][v] = 0.f;

  for (int k0 = 0; k0 < K; k0 += 32) {
    v16h af[4], bf[2];
#pragma unroll
    for (int mi = 0; mi < 4; ++mi) {
      int r = m0 + mi * 16 + sub; r = (r > aclamp) ? aclamp : r;
      const h16* p = A + (size_t)r * lda + k0 + koff;
      v8h lo = *(const v8h*)p;
      v8h hh = *(const v8h*)(p + 16);
#pragma unroll
      for (int j = 0; j < 8; ++j) { af[mi][j] = lo[j]; af[mi][8 + j] = hh[j]; }
    }
#pragma unroll
    for (int ni = 0; ni < 2; ++ni) {
      int r = n0 + ni * 16 + sub; r = (r > bclamp) ? bclamp : r;
      const h16* p = Bt + (size_t)r * ldb + k0 + koff;
      v8h lo = *(const v8h*)p;
      v8h hh = *(const v8h*)(p + 16);
#pragma unroll
      for (int j = 0; j < 8; ++j) { bf[ni][j] = lo[j]; bf[ni][8 + j] = hh[j]; }
    }
#pragma unroll
    for (int mi = 0; mi < 4; ++mi)
#pragma unroll
      for (int ni = 0; ni < 2; ++ni)
        acc[mi][ni] = __builtin_amdgcn_wmma_f32_16x16x32_f16(
            false, af[mi], false, bf[ni], (short)0, acc[mi][ni], false, false);
  }

#pragma unroll
  for (int mi = 0; mi < 4; ++mi)
#pragma unroll
    for (int ni = 0; ni < 2; ++ni) {
      int c = n0 + ni * 16 + sub;
      if (c >= N) continue;
      float bv = bias ? bias[c] : 0.f;
#pragma unroll
      for (int v = 0; v < 8; ++v) {
        int r = m0 + mi * 16 + hi * 8 + v;
        if (r >= M) continue;
        float val = acc[mi][ni][v] + bv;
        if (ep == 1) val = val > 0.f ? val : 0.2f * val;
        C[(size_t)r * ldc + c] = val;
      }
    }
}

// ------------------------------ prep kernels -------------------------------
// conv weight (O,C,taps) f32 -> Bt f16 (O, Kpad), col = tap*C + c
__global__ void k_prep_convw(const float* __restrict__ src, h16* __restrict__ dst,
                             int O, int C, int taps, int Kpad) {
  size_t idx = (size_t)blockIdx.x * 256 + threadIdx.x;
  if (idx >= (size_t)O * Kpad) return;
  int o = (int)(idx / Kpad), kk = (int)(idx % Kpad);
  float v = 0.f;
  if (kk < taps * C) {
    int tap = kk / C, c = kk % C;
    v = src[((size_t)o * C + c) * taps + tap];
  }
  dst[idx] = (h16)v;
}

// linear weight (O,Kin) f32 -> Bt f16 (Opad, Kpad), zero padded
__global__ void k_prep_linw(const float* __restrict__ src, h16* __restrict__ dst,
                            int O, int Opad, int Kin, int Kpad) {
  size_t idx = (size_t)blockIdx.x * 256 + threadIdx.x;
  if (idx >= (size_t)Opad * Kpad) return;
  int o = (int)(idx / Kpad), k = (int)(idx % Kpad);
  float v = (o < O && k < Kin) ? src[(size_t)o * Kin + k] : 0.f;
  dst[idx] = (h16)v;
}

// content fused weight: Wc[h][m] = sum_k dct[m][k]*cl_w[h][k]; (256 x 96, m<80)
__global__ void k_prep_contentw(const float* __restrict__ dct, const float* __restrict__ clw,
                                h16* __restrict__ dst) {
  int idx = blockIdx.x * 256 + threadIdx.x;
  if (idx >= 256 * 96) return;
  int h = idx / 96, m = idx % 96;
  float s = 0.f;
  if (m < 80)
    for (int k = 0; k < 40; ++k) s += dct[m * 40 + k] * clw[h * 40 + k];
  dst[idx] = (h16)s;
}

__global__ void k_cbnorm(const float* __restrict__ cb, float* __restrict__ cbn) {
  int k = blockIdx.x * 256 + threadIdx.x;
  if (k >= CBK) return;
  float s = 0.f;
  for (int d = 0; d < CBD; ++d) { float v = cb[k * CBD + d]; s += v * v; }
  float inv = 1.f / fmaxf(sqrtf(s), 1e-12f);
  for (int d = 0; d < CBD; ++d) cbn[k * CBD + d] = cb[k * CBD + d] * inv;
}

// ------------------------------ staging kernels ----------------------------
// im2col from (B,T,C) f32 activations -> f16 (BT, Kpad); taps=5 pad=2 or taps=1
__global__ void k_im2col_btc(const float* __restrict__ src, h16* __restrict__ dst,
                             int C, int taps, int Kpad) {
  size_t idx = (size_t)blockIdx.x * 256 + threadIdx.x;
  if (idx >= (size_t)BT * Kpad) return;
  int bt = (int)(idx / Kpad), kk = (int)(idx % Kpad);
  int b = bt >> 10, t = bt & (TT - 1);
  float v = 0.f;
  if (kk < taps * C) {
    int tap = kk / C, c = kk % C, tt = t + tap - taps / 2;
    if (tt >= 0 && tt < TT) v = src[((size_t)b * TT + tt) * C + c];
  }
  dst[idx] = (h16)v;
}

// im2col from (B,C,T) f32 inputs -> f16 (BT, Kpad)
__global__ void k_im2col_bct(const float* __restrict__ src, h16* __restrict__ dst,
                             int C, int taps, int Kpad) {
  size_t idx = (size_t)blockIdx.x * 256 + threadIdx.x;
  if (idx >= (size_t)BT * Kpad) return;
  int bt = (int)(idx / Kpad), kk = (int)(idx % Kpad);
  int b = bt >> 10, t = bt & (TT - 1);
  float v = 0.f;
  if (kk < taps * C) {
    int tap = kk / C, c = kk % C, tt = t + tap - taps / 2;
    if (tt >= 0 && tt < TT) v = src[((size_t)b * C + c) * TT + tt];
  }
  dst[idx] = (h16)v;
}

// f32 (rows,Csrc) -> f16 (rows,Kpad) zero-padded
__global__ void k_cvt_f16(const float* __restrict__ src, h16* __restrict__ dst,
                          int Csrc, int Kpad, size_t total) {
  size_t idx = (size_t)blockIdx.x * 256 + threadIdx.x;
  if (idx >= total) return;
  int r = (int)(idx / Kpad), k = (int)(idx % Kpad);
  dst[idx] = (k < Csrc) ? (h16)src[(size_t)r * Csrc + k] : (h16)0.f;
}

__global__ void k_fill(float* __restrict__ p, float v, size_t n) {
  size_t i = (size_t)blockIdx.x * 256 + threadIdx.x;
  if (i < n) p[i] = v;
}

// ------------------------------ WaveNet pieces -----------------------------
__global__ void k_gate(const float* __restrict__ xin, h16* __restrict__ acts) {
  size_t idx = (size_t)blockIdx.x * 256 + threadIdx.x;
  if (idx >= (size_t)BT * HH) return;
  int row = (int)(idx >> 8), h = (int)(idx & (HH - 1));
  float a = xin[(size_t)row * 512 + h];
  float g = xin[(size_t)row * 512 + HH + h];
  acts[idx] = (h16)(tanhf(a) * (1.f / (1.f + expf(-g))));
}

__global__ void k_resskip(float* __restrict__ x, float* __restrict__ skip,
                          const float* __restrict__ rs, int addres) {
  size_t idx = (size_t)blockIdx.x * 256 + threadIdx.x;
  if (idx >= (size_t)BT * HH) return;
  int row = (int)(idx >> 8), h = (int)(idx & (HH - 1));
  float rlo = rs[(size_t)row * 512 + h];
  float rhi = rs[(size_t)row * 512 + HH + h];
  if (addres) x[idx] += rlo;
  skip[idx] += rhi;
}

__global__ __launch_bounds__(256) void k_meanpool(const float* __restrict__ src,
                                                  float* __restrict__ dst) {
  int b = blockIdx.x, h = threadIdx.x;
  float s = 0.f;
  for (int t = 0; t < TT; ++t) s += src[((size_t)b * TT + t) * HH + h];
  dst[b * HH + h] = s * (1.f / TT);
}

// ------------------------------ RVQ pieces ---------------------------------
// one thread per (b,t): cosine argmax over 1024 codes (normalized cb in LDS),
// gathers selected raw code row into afrag f16 (BT,32), accumulates MSE loss.
__global__ __launch_bounds__(256) void k_rvq_nn(const float* __restrict__ ze,
                                                const float* __restrict__ cbn,
                                                const float* __restrict__ cb,
                                                h16* __restrict__ afrag,
                                                float* __restrict__ loss_slot) {
  __shared__ float cbs[CBK * CBD];
  for (int i = threadIdx.x; i < CBK * CBD; i += 256) cbs[i] = cbn[i];
  __syncthreads();
  int bt = blockIdx.x * 256 + threadIdx.x;
  float z[CBD], zn[CBD];
  float nrm = 0.f;
#pragma unroll
  for (int d = 0; d < CBD; ++d) { z[d] = ze[(size_t)bt * CBD + d]; nrm += z[d] * z[d]; }
  nrm = 1.f / fmaxf(sqrtf(nrm), 1e-12f);
#pragma unroll
  for (int d = 0; d < CBD; ++d) zn[d] = z[d] * nrm;
  int bestk = 0; float best = -1e30f;
  for (int k = 0; k < CBK; ++k) {
    const float* c = &cbs[k * CBD];
    float s = 0.f;
#pragma unroll
    for (int d = 0; d < CBD; ++d) s += zn[d] * c[d];
    if (s > best) { best = s; bestk = k; }
  }
  const float* cr = cb + (size_t)bestk * CBD;
  h16* ap = afrag + (size_t)bt * 32;
  float l = 0.f;
#pragma unroll
  for (int d = 0; d < CBD; ++d) {
    float cv = cr[d];
    ap[d] = (h16)cv;
    float df = z[d] - cv;
    l += df * df;
  }
#pragma unroll
  for (int d = CBD; d < 32; ++d) ap[d] = (h16)0.f;
  atomicAdd(loss_slot, l);
}

__global__ void k_add_inplace(float* __restrict__ a, const float* __restrict__ b, size_t n) {
  size_t i = (size_t)blockIdx.x * 256 + threadIdx.x;
  if (i < n) a[i] += b[i];
}
__global__ void k_sub_inplace(float* __restrict__ a, const float* __restrict__ b, size_t n) {
  size_t i = (size_t)blockIdx.x * 256 + threadIdx.x;
  if (i < n) a[i] -= b[i];
}
__global__ void k_sub2(float* __restrict__ a, const float* __restrict__ b,
                       const float* __restrict__ c, size_t n) {
  size_t i = (size_t)blockIdx.x * 256 + threadIdx.x;
  if (i < n) a[i] = a[i] - b[i] - c[i];
}

// ------------------------------ output pieces ------------------------------
// batched matrix transpose: src (b,R,C) -> dst (b,C,R)
__global__ void k_transpose(const float* __restrict__ src, float* __restrict__ dst,
                            int R, int C) {
  __shared__ float tile[32][33];
  int b = blockIdx.z;
  const float* s = src + (size_t)b * R * C;
  float* d = dst + (size_t)b * R * C;
  int c0 = blockIdx.x * 32, r0 = blockIdx.y * 32;
  for (int i = threadIdx.y; i < 32; i += 8) {
    int r = r0 + i, c = c0 + threadIdx.x;
    tile[i][threadIdx.x] = (r < R && c < C) ? s[(size_t)r * C + c] : 0.f;
  }
  __syncthreads();
  for (int i = threadIdx.y; i < 32; i += 8) {
    int c = c0 + i, r = r0 + threadIdx.x;
    if (c < C && r < R) d[(size_t)c * R + r] = tile[threadIdx.x][i];
  }
}

// per (b,t): outs = zp+zc+zr*mask; LN over D; *gamma +beta; store (B,D,T)
__global__ __launch_bounds__(256) void k_adain(const float* __restrict__ zp,
                                               const float* __restrict__ zc,
                                               const float* __restrict__ zr,
                                               const float* __restrict__ res_mask,
                                               const float* __restrict__ style,
                                               float* __restrict__ out) {
  __shared__ float shs[256], shq[256];
  int bt = blockIdx.x, b = bt >> 10, t = bt & (TT - 1);
  int tid = threadIdx.x;
  float m = res_mask[b];
  float v[4], s = 0.f, s2 = 0.f;
#pragma unroll
  for (int j = 0; j < 4; ++j) {
    int d = tid + j * 256;
    size_t i = (size_t)bt * DD + d;
    float x = zp[i] + zc[i] + zr[i] * m;
    v[j] = x; s += x; s2 += x * x;
  }
  shs[tid] = s; shq[tid] = s2;
  __syncthreads();
  for (int off = 128; off > 0; off >>= 1) {
    if (tid < off) { shs[tid] += shs[tid + off]; shq[tid] += shq[tid + off]; }
    __syncthreads();
  }
  float mean = shs[0] * (1.f / DD);
  float var = shq[0] * (1.f / DD) - mean * mean;
  float inv = rsqrtf(var + 1e-5f);
#pragma unroll
  for (int j = 0; j < 4; ++j) {
    int d = tid + j * 256;
    float g = style[(size_t)b * 2048 + d];
    float be = style[(size_t)b * 2048 + DD + d];
    float val = (v[j] - mean) * inv * g + be;
    out[((size_t)b * DD + d) * TT + t] = val;
  }
}

__global__ void k_finalize(const float* __restrict__ loss, float* __restrict__ oc,
                           float* __restrict__ ocb) {
  float s = 0.f;
  for (int i = 0; i < 6; ++i) s += loss[i];
  s *= (1.f / (float)(BB * CBD * TT));
  *oc = s; *ocb = s;
}

// ===========================================================================
extern "C" void kernel_launch(void* const* d_in, const int* in_sizes, int n_in,
                              void* d_out, int out_size, void* d_ws, size_t ws_size,
                              hipStream_t stream) {
  (void)in_sizes; (void)n_in; (void)out_size; (void)ws_size;
  // ---- inputs (setup_inputs order) ----
  const float* x        = (const float*)d_in[0];
  const float* prosody  = (const float*)d_in[1];
  const float* melseg   = (const float*)d_in[2];
  const float* res_mask = (const float*)d_in[5];
  const float* dct      = (const float*)d_in[6];
  const float* ml_w = (const float*)d_in[7];   const float* ml_b = (const float*)d_in[8];
  const float* mwn_in_w = (const float*)d_in[9];  const float* mwn_in_b = (const float*)d_in[10];
  const float* mwn_rs_w = (const float*)d_in[11]; const float* mwn_rs_b = (const float*)d_in[12];
  const float* ml2_w = (const float*)d_in[13]; const float* ml2_b = (const float*)d_in[14];
  const float* cl_w  = (const float*)d_in[15]; const float* cl_b  = (const float*)d_in[16];
  const float* cwn_in_w = (const float*)d_in[17]; const float* cwn_in_b = (const float*)d_in[18];
  const float* cwn_rs_w = (const float*)d_in[19]; const float* cwn_rs_b = (const float*)d_in[20];
  const float* cl2_w = (const float*)d_in[21]; const float* cl2_b = (const float*)d_in[22];
  const float* rvq_in_w[6]  = { (const float*)d_in[23],
                                (const float*)d_in[28], (const float*)d_in[28] + 8*1024,
                                (const float*)d_in[33], (const float*)d_in[33] + 8*1024,
                                (const float*)d_in[33] + 2*8*1024 };
  const float* rvq_in_b[6]  = { (const float*)d_in[24],
                                (const float*)d_in[29], (const float*)d_in[29] + 8,
                                (const float*)d_in[34], (const float*)d_in[34] + 8,
                                (const float*)d_in[34] + 16 };
  const float* rvq_cb[6]    = { (const float*)d_in[25],
                                (const float*)d_in[30], (const float*)d_in[30] + CBK*CBD,
                                (const float*)d_in[35], (const float*)d_in[35] + CBK*CBD,
                                (const float*)d_in[35] + 2*CBK*CBD };
  const float* rvq_out_w[6] = { (const float*)d_in[26],
                                (const float*)d_in[31], (const float*)d_in[31] + 1024*8,
                                (const float*)d_in[36], (const float*)d_in[36] + 1024*8,
                                (const float*)d_in[36] + 2*1024*8 };
  const float* rvq_out_b[6] = { (const float*)d_in[27],
                                (const float*)d_in[32], (const float*)d_in[32] + 1024,
                                (const float*)d_in[37], (const float*)d_in[37] + 1024,
                                (const float*)d_in[37] + 2*1024 };
  const float* se_w1 = (const float*)d_in[38]; const float* se_b1 = (const float*)d_in[39];
  const float* se_w2 = (const float*)d_in[40]; const float* se_b2 = (const float*)d_in[41];
  const float* se_wo = (const float*)d_in[42]; const float* se_bo = (const float*)d_in[43];
  const float* tl_w  = (const float*)d_in[44]; const float* tl_b  = (const float*)d_in[45];

  // ---- output regions ----
  const size_t OSZ = (size_t)BB * DD * TT;
  float* out_o      = (float*)d_out;
  float* out_zp     = out_o + OSZ;
  float* out_zc     = out_zp + OSZ;
  float* out_zr     = out_zc + OSZ;
  float* out_commit = out_zr + OSZ;
  float* out_cb     = out_commit + 1;
  float* out_timbre = out_cb + 1;          // (B,1024)

  // ---- workspace carve ----
  char* wsb = (char*)d_ws;
  size_t off = 0;
  auto alloc = [&](size_t bytes) -> void* {
    void* p = wsb + off;
    off = (off + bytes + 255) & ~(size_t)255;
    return p;
  };
  h16*   afrag   = (h16*)  alloc((size_t)BT * 1280 * 2);   // generic f16 A staging
  float* xinbuf  = (float*)alloc((size_t)BT * 512 * 4);
  float* rsbuf   = (float*)alloc((size_t)BT * 512 * 4);
  float* wxbuf   = (float*)alloc((size_t)BT * HH * 4);
  float* skipbuf = (float*)alloc((size_t)BT * HH * 4);
  h16*   acts16  = (h16*)  alloc((size_t)BT * HH * 2);
  float* f0buf   = (float*)alloc((size_t)BT * DD * 4);
  float* contbuf = (float*)alloc((size_t)BT * DD * 4);
  float* xbtc    = (float*)alloc((size_t)BT * DD * 4);
  float* zp      = (float*)alloc((size_t)BT * DD * 4);
  float* zc      = (float*)alloc((size_t)BT * DD * 4);
  float* zr      = (float*)alloc((size_t)BT * DD * 4);
  float* tbuf    = (float*)alloc((size_t)BT * DD * 4);
  float* zebuf   = (float*)alloc((size_t)BT * CBD * 4);
  float* stylebuf= (float*)alloc((size_t)BB * 2048 * 4);
  float* pooled  = (float*)alloc((size_t)BB * HH * 4);
  float* lossbuf = (float*)alloc(8 * 4);
  h16* Wse1 = (h16*)alloc((size_t)256 * 416 * 2);
  h16* Wse2 = (h16*)alloc((size_t)256 * 1280 * 2);
  h16* Wseo = (h16*)alloc((size_t)1024 * 256 * 2);
  h16* Wml  = (h16*)alloc((size_t)256 * 32 * 2);
  h16* Wml2 = (h16*)alloc((size_t)1024 * 256 * 2);
  h16* Wc   = (h16*)alloc((size_t)256 * 96 * 2);
  h16* Wcl2 = (h16*)alloc((size_t)1024 * 256 * 2);
  h16* Wtl  = (h16*)alloc((size_t)2048 * 1024 * 2);
  h16* Wmin = (h16*)alloc((size_t)8 * 512 * 1280 * 2);
  h16* Wmrs = (h16*)alloc((size_t)8 * 512 * 256 * 2);
  h16* Wcin = (h16*)alloc((size_t)16 * 512 * 1280 * 2);
  h16* Wcrs = (h16*)alloc((size_t)16 * 512 * 256 * 2);
  h16*   Win[6]; h16* Wout[6]; float* cbn[6];
  for (int i = 0; i < 6; ++i) {
    Win[i]  = (h16*)alloc((size_t)16 * 1024 * 2);
    Wout[i] = (h16*)alloc((size_t)1024 * 32 * 2);
    cbn[i]  = (float*)alloc((size_t)CBK * CBD * 4);
  }

  auto NB = [](size_t n) { return dim3((unsigned)((n + 255) / 256)); };
  auto gemm = [&](const h16* A, int lda, const h16* Bt, int ldb, int bclamp,
                  const float* bias, float* C, int ldc, int M, int N, int K, int ep) {
    dim3 g((N + 127) / 128, (M + 127) / 128);
    gemm_wmma_nt<<<g, 256, 0, stream>>>(A, lda, M - 1, Bt, ldb, bclamp, bias, C, ldc, M, N, K, ep);
  };
  auto fill0 = [&](float* p, size_t n) { k_fill<<<NB(n), 256, 0, stream>>>(p, 0.f, n); };

  // ================= phase 0: weight prep =================
  k_prep_convw<<<NB((size_t)256 * 416), 256, 0, stream>>>(se_w1, Wse1, 256, 80, 5, 416);
  k_prep_convw<<<NB((size_t)256 * 1280), 256, 0, stream>>>(se_w2, Wse2, 256, 256, 5, 1280);
  k_prep_linw<<<NB((size_t)1024 * 256), 256, 0, stream>>>(se_wo, Wseo, 1024, 1024, 256, 256);
  k_prep_linw<<<NB((size_t)256 * 32), 256, 0, stream>>>(ml_w, Wml, 256, 256, 20, 32);
  k_prep_linw<<<NB((size_t)1024 * 256), 256, 0, stream>>>(ml2_w, Wml2, 1024, 1024, 256, 256);
  k_prep_contentw<<<NB((size_t)256 * 96), 256, 0, stream>>>(dct, cl_w, Wc);
  k_prep_linw<<<NB((size_t)1024 * 256), 256, 0, stream>>>(cl2_w, Wcl2, 1024, 1024, 256, 256);
  k_prep_linw<<<NB((size_t)2048 * 1024), 256, 0, stream>>>(tl_w, Wtl, 2048, 2048, 1024, 1024);
  for (int i = 0; i < 8; ++i) {
    k_prep_convw<<<NB((size_t)512 * 1280), 256, 0, stream>>>(
        mwn_in_w + (size_t)i * 512 * 256 * 5, Wmin + (size_t)i * 512 * 1280, 512, 256, 5, 1280);
    k_prep_linw<<<NB((size_t)512 * 256), 256, 0, stream>>>(
        mwn_rs_w + (size_t)i * 512 * 256, Wmrs + (size_t)i * 512 * 256, 512, 512, 256, 256);
  }
  for (int i = 0; i < 16; ++i) {
    k_prep_convw<<<NB((size_t)512 * 1280), 256, 0, stream>>>(
        cwn_in_w + (size_t)i * 512 * 256 * 5, Wcin + (size_t)i * 512 * 1280, 512, 256, 5, 1280);
    k_prep_linw<<<NB((size_t)512 * 256), 256, 0, stream>>>(
        cwn_rs_w + (size_t)i * 512 * 256, Wcrs + (size_t)i * 512 * 256, 512, 512, 256, 256);
  }
  for (int i = 0; i < 6; ++i) {
    k_prep_linw<<<NB((size_t)16 * 1024), 256, 0, stream>>>(rvq_in_w[i], Win[i], 8, 16, 1024, 1024);
    k_prep_linw<<<NB((size_t)1024 * 32), 256, 0, stream>>>(rvq_out_w[i], Wout[i], 1024, 1024, 8, 32);
    k_cbnorm<<<NB(CBK), 256, 0, stream>>>(rvq_cb[i], cbn[i]);
  }
  fill0(lossbuf, 8);

  // ================= phase 1: style encoder =================
  k_im2col_bct<<<NB((size_t)BT * 416), 256, 0, stream>>>(melseg, afrag, 80, 5, 416);
  gemm(afrag, 416, Wse1, 416, 255, se_b1, wxbuf, 256, BT, 256, 416, 1);     // h1 (leaky)
  k_im2col_btc<<<NB((size_t)BT * 1280), 256, 0, stream>>>(wxbuf, afrag, 256, 5, 1280);
  gemm(afrag, 1280, Wse2, 1280, 255, se_b2, skipbuf, 256, BT, 256, 1280, 1); // h2 (leaky)
  k_meanpool<<<dim3(BB), 256, 0, stream>>>(skipbuf, pooled);
  k_cvt_f16<<<NB((size_t)BB * 256), 256, 0, stream>>>(pooled, afrag, 256, 256, (size_t)BB * 256);
  gemm(afrag, 256, Wseo, 256, 1023, se_bo, out_timbre, 1024, BB, 1024, 256, 0); // timbre

  // ================= phase 2: prosody branch =================
  k_im2col_bct<<<NB((size_t)BT * 32), 256, 0, stream>>>(prosody, afrag, 20, 1, 32);
  gemm(afrag, 32, Wml, 32, 255, ml_b, wxbuf, 256, BT, 256, 32, 0);
  fill0(skipbuf, (size_t)BT * HH);
  for (int i = 0; i < 8; ++i) {
    k_im2col_btc<<<NB((size_t)BT * 1280), 256, 0, stream>>>(wxbuf, afrag, 256, 5, 1280);
    gemm(afrag, 1280, Wmin + (size_t)i * 512 * 1280, 1280, 511,
         mwn_in_b + (size_t)i * 512, xinbuf, 512, BT, 512, 1280, 0);
    k_gate<<<NB((size_t)BT * HH), 256, 0, stream>>>(xinbuf, acts16);
    gemm(acts16, 256, Wmrs + (size_t)i * 512 * 256, 256, 511,
         mwn_rs_b + (size_t)i * 512, rsbuf, 512, BT, 512, 256, 0);
    k_resskip<<<NB((size_t)BT * HH), 256, 0, stream>>>(wxbuf, skipbuf, rsbuf, (i < 7) ? 1 : 0);
  }
  k_cvt_f16<<<NB((size_t)BT * 256), 256, 0, stream>>>(skipbuf, afrag, 256, 256, (size_t)BT * 256);
  gemm(afrag, 256, Wml2, 256, 1023, ml2_b, f0buf, 1024, BT, 1024, 256, 0);

  // ================= phase 3: content branch =================
  k_im2col_bct<<<NB((size_t)BT * 96), 256, 0, stream>>>(melseg, afrag, 80, 1, 96);
  gemm(afrag, 96, Wc, 96, 255, cl_b, wxbuf, 256, BT, 256, 96, 0);
  fill0(skipbuf, (size_t)BT * HH);
  for (int i = 0; i < 16; ++i) {
    k_im2col_btc<<<NB((size_t)BT * 1280), 256, 0, stream>>>(wxbuf, afrag, 256, 5, 1280);
    gemm(afrag, 1280, Wcin + (size_t)i * 512 * 1280, 1280, 511,
         cwn_in_b + (size_t)i * 512, xinbuf, 512, BT, 512, 1280, 0);
    k_gate<<<NB((size_t)BT * HH), 256, 0, stream>>>(xinbuf, acts16);
    gemm(acts16, 256, Wcrs + (size_t)i * 512 * 256, 256, 511,
         cwn_rs_b + (size_t)i * 512, rsbuf, 512, BT, 512, 256, 0);
    k_resskip<<<NB((size_t)BT * HH), 256, 0, stream>>>(wxbuf, skipbuf, rsbuf, (i < 15) ? 1 : 0);
  }
  k_cvt_f16<<<NB((size_t)BT * 256), 256, 0, stream>>>(skipbuf, afrag, 256, 256, (size_t)BT * 256);
  gemm(afrag, 256, Wcl2, 256, 1023, cl2_b, contbuf, 1024, BT, 1024, 256, 0);

  // ================= phase 4-6: RVQ =================
  const size_t NBD = (size_t)BT * DD;
  auto rvq_step = [&](float* resid, int cbi, int slot, float* target, bool update_resid) {
    k_cvt_f16<<<NB(NBD), 256, 0, stream>>>(resid, afrag, 1024, 1024, NBD);
    gemm(afrag, 1024, Win[cbi], 1024, 15, rvq_in_b[cbi], zebuf, CBD, BT, CBD, 1024, 0);
    k_rvq_nn<<<dim3(BT / 256), 256, 0, stream>>>(zebuf, cbn[cbi], rvq_cb[cbi], afrag,
                                                 lossbuf + slot);
    gemm(afrag, 32, Wout[cbi], 32, 1023, rvq_out_b[cbi], target, 1024, BT, 1024, 32, 0);
    if (update_resid) k_sub_inplace<<<NB(NBD), 256, 0, stream>>>(resid, target, NBD);
  };
  // prosody RVQ (n_q = 1)
  rvq_step(f0buf, 0, 0, zp, false);
  // content RVQ (n_q = 2)
  rvq_step(contbuf, 1, 1, zc, true);
  rvq_step(contbuf, 2, 2, tbuf, false);
  k_add_inplace<<<NB(NBD), 256, 0, stream>>>(zc, tbuf, NBD);
  // residual feature + residual RVQ (n_q = 3)
  k_transpose<<<dim3(TT / 32, DD / 32, BB), dim3(32, 8), 0, stream>>>(x, xbtc, DD, TT);
  k_sub2<<<NB(NBD), 256, 0, stream>>>(xbtc, zp, zc, NBD);
  rvq_step(xbtc, 3, 3, zr, true);
  rvq_step(xbtc, 4, 4, tbuf, true);
  k_add_inplace<<<NB(NBD), 256, 0, stream>>>(zr, tbuf, NBD);
  rvq_step(xbtc, 5, 5, tbuf, false);
  k_add_inplace<<<NB(NBD), 256, 0, stream>>>(zr, tbuf, NBD);

  // ================= phase 7: AdaIN + outputs =================
  k_cvt_f16<<<NB((size_t)BB * 1024), 256, 0, stream>>>(out_timbre, afrag, 1024, 1024,
                                                       (size_t)BB * 1024);
  gemm(afrag, 1024, Wtl, 1024, 2047, tl_b, stylebuf, 2048, BB, 2048, 1024, 0);
  k_adain<<<dim3(BT), 256, 0, stream>>>(zp, zc, zr, res_mask, stylebuf, out_o);
  k_transpose<<<dim3(DD / 32, TT / 32, BB), dim3(32, 8), 0, stream>>>(zp, out_zp, TT, DD);
  k_transpose<<<dim3(DD / 32, TT / 32, BB), dim3(32, 8), 0, stream>>>(zc, out_zc, TT, DD);
  k_transpose<<<dim3(DD / 32, TT / 32, BB), dim3(32, 8), 0, stream>>>(zr, out_zr, TT, DD);
  k_finalize<<<1, 1, 0, stream>>>(lossbuf, out_commit, out_cb);
}
